// CustomTransformerBlock_85289460564162
// MI455X (gfx1250) — matrix operand order repaired and checked by reading, not compiled, not run on hardware
//
#include <hip/hip_runtime.h>
#include <hip/hip_bf16.h>
#include <math.h>

// ---------------------------------------------------------------------------
// MI455X (gfx1250) transformer block: all GEMMs via v_wmma_f32_16x16x32_bf16,
// flash attention (no S x S materialization), bf16 data path, f32 accumulate.
// GEMM: 32x64 per wave (2x4 accumulators) -> 1.5 b128 loads per WMMA, with
// enough VGPR headroom for the compiler to unroll/software-pipeline the K loop.
// ---------------------------------------------------------------------------

#define E_   1024
#define H_   16
#define D_   64
#define MLP_ 4096
#define S_   2048
#define B_   4
#define M_   (B_ * S_)   // 8192 rows of activations

typedef __bf16 bf16;
typedef __attribute__((ext_vector_type(16))) __bf16 v16bf;
typedef __attribute__((ext_vector_type(8)))  float  v8f;
typedef __attribute__((ext_vector_type(4)))  unsigned int u32x4;
typedef __attribute__((ext_vector_type(4)))  __bf16 bf16x4;

union Frag {
  v16bf v;
  u32x4 q[2];
};

// A-matrix 16x32 bf16 fragment (ISA 7.12.2): lane m = lane&15,
// k-base = 8*(lane>=16); elements 0..7 -> k = kb..kb+7, 8..15 -> kb+16..kb+23.
__device__ __forceinline__ v16bf load_a_frag(const bf16* A, int lda, int m0, int k0) {
  int lane = threadIdx.x & 31;
  int m  = m0 + (lane & 15);
  int kb = k0 + ((lane >> 4) << 3);
  const bf16* p = A + (size_t)m * lda + kb;
  Frag f;
  f.q[0] = *(const u32x4*)p;
  f.q[1] = *(const u32x4*)(p + 16);
  return f.v;
}

// B-matrix 32x16 bf16 fragment: lane n = n0 + (lane&15),
// k-base = 8*(lane>=16); elements 0..7 -> kb..kb+7, 8..15 -> kb+16..kb+23. W row-major [N,K] -> two b128 loads.
__device__ __forceinline__ v16bf load_b_frag(const bf16* W, int ldw, int n0, int k0) {
  int lane = threadIdx.x & 31;
  int n  = n0 + (lane & 15);
  int kb = k0 + ((lane >> 4) << 3);
  const bf16* p = W + (size_t)n * ldw + kb;
  Frag f;
  f.q[0] = *(const u32x4*)p;
  f.q[1] = *(const u32x4*)(p + 16);
  return f.v;
}

__device__ __forceinline__ v8f wmma_bf16(v16bf a, v16bf b, v8f c) {
  return __builtin_amdgcn_wmma_f32_16x16x32_bf16(false, a, false, b, (short)0, c,
                                                 false, false);
}

// ---------------------------------------------------------------------------
// Elementwise fp32 -> bf16 weight conversion (one-time; bf16 weights = 24 MB,
// resident in 192 MB L2 for all subsequent GEMMs).
// ---------------------------------------------------------------------------
__global__ void __launch_bounds__(256)
cvt_bf16_kernel(const float* __restrict__ in, bf16* __restrict__ out, int n) {
  int i = (blockIdx.x * 256 + threadIdx.x) * 4;
  if (i + 3 < n) {
    float4 v = *(const float4*)(in + i);
    bf16x4 o;
    o[0] = (bf16)v.x; o[1] = (bf16)v.y; o[2] = (bf16)v.z; o[3] = (bf16)v.w;
    *(bf16x4*)(out + i) = o;
  }
}

// ---------------------------------------------------------------------------
// LayerNorm over E=1024, one row per 256-thread block (8 waves), bf16 output.
// ---------------------------------------------------------------------------
__global__ void __launch_bounds__(256)
layernorm_kernel(const float* __restrict__ x, const float* __restrict__ g,
                 const float* __restrict__ bta, bf16* __restrict__ out) {
  __shared__ float red[2][8];
  int row = blockIdx.x;
  int t = threadIdx.x;
  const float* xr = x + (size_t)row * E_;
  float4 v = *(const float4*)(xr + t * 4);
  float s  = v.x + v.y + v.z + v.w;
  float ss = v.x * v.x + v.y * v.y + v.z * v.z + v.w * v.w;
#pragma unroll
  for (int m = 1; m < 32; m <<= 1) {
    s  += __shfl_xor(s, m);
    ss += __shfl_xor(ss, m);
  }
  int wave = t >> 5, lane = t & 31;
  if (lane == 0) { red[0][wave] = s; red[1][wave] = ss; }
  __syncthreads();
  float tots = 0.f, totss = 0.f;
#pragma unroll
  for (int i = 0; i < 8; ++i) { tots += red[0][i]; totss += red[1][i]; }
  float mu  = tots * (1.0f / E_);
  float var = totss * (1.0f / E_) - mu * mu;
  float rs  = rsqrtf(var + 1e-5f);
  float4 gv = *(const float4*)(g + t * 4);
  float4 bv = *(const float4*)(bta + t * 4);
  bf16x4 o;
  o[0] = (bf16)((v.x - mu) * rs * gv.x + bv.x);
  o[1] = (bf16)((v.y - mu) * rs * gv.y + bv.y);
  o[2] = (bf16)((v.z - mu) * rs * gv.z + bv.z);
  o[3] = (bf16)((v.w - mu) * rs * gv.w + bv.w);
  *(bf16x4*)(out + (size_t)row * E_ + t * 4) = o;
}

// ---------------------------------------------------------------------------
// Generic WMMA GEMM: C[M,N] = A[M,K] @ W[N,K]^T.
// 128 threads = 4 waves (2x2), block tile 64x128, wave tile 32x64
// (2x4 f32 accumulators): per 32-K step, 8 WMMAs for 12 b128 loads.
// One global_prefetch_b8 per K-step for the streamed A operand.
// MODE 0: QKV scatter epilogue (Q,K -> [B,H,S,D] bf16; V -> [B,H,D,S] bf16)
// MODE 1: + bias + residual -> f32
// MODE 2: + bias + exact GELU -> bf16
// ---------------------------------------------------------------------------
template <int MODE>
__global__ void __launch_bounds__(128)
gemm_kernel(const bf16* __restrict__ A, const bf16* __restrict__ W,
            int Ndim, int Kdim,
            const float* __restrict__ bias, const float* __restrict__ resid,
            float* __restrict__ outF, bf16* __restrict__ outB,
            bf16* __restrict__ Qd, bf16* __restrict__ Kd, bf16* __restrict__ Vt) {
  int wave = threadIdx.x >> 5;
  int lane = threadIdx.x & 31;
  int m0 = blockIdx.y * 64 + (wave >> 1) * 32;
  int n0 = blockIdx.x * 128 + (wave & 1) * 64;

  v8f acc[2][4] = {};
  for (int k0 = 0; k0 < Kdim; k0 += 32) {
    v16bf a[2], b[4];
#pragma unroll
    for (int i = 0; i < 2; ++i) a[i] = load_a_frag(A, Kdim, m0 + 16 * i, k0);
#pragma unroll
    for (int j = 0; j < 4; ++j) b[j] = load_b_frag(W, Kdim, n0 + 16 * j, k0);
    if (k0 + 32 < Kdim) {  // gfx1250 global_prefetch_b8: next K-step of A
      const bf16* pa =
          A + (size_t)(m0 + (lane & 15) + ((lane >> 4) << 4)) * Kdim + k0 + 32;
      __builtin_prefetch(pa, 0, 0);
    }
#pragma unroll
    for (int i = 0; i < 2; ++i)
#pragma unroll
      for (int j = 0; j < 4; ++j)
        acc[i][j] = wmma_bf16(a[i], b[j], acc[i][j]);
  }

  int rofs = (lane >> 4) << 3;   // C/D layout: lanes 16..31 hold rows m+8
  int nlo  = lane & 15;

  if (MODE == 0) {
    // q/k/v selector, head, and d-base are constant within each aligned
    // 16-wide n-tile (boundaries are 64-aligned) -> hoist out of m/r loops.
#pragma unroll
    for (int tj = 0; tj < 4; ++tj) {
      int nt = n0 + tj * 16;
      int which = nt >> 10;
      int h = (nt & 1023) >> 6;
      int d = (nt & 63) + nlo;
#pragma unroll
      for (int ti = 0; ti < 2; ++ti)
#pragma unroll
        for (int r = 0; r < 8; ++r) {
          int m = m0 + ti * 16 + r + rofs;
          int b = m >> 11, s = m & 2047;
          size_t bh = (size_t)b * H_ + h;
          bf16 bv = (bf16)acc[ti][tj][r];
          if (which == 0)      Qd[(bh * S_ + s) * D_ + d] = bv;
          else if (which == 1) Kd[(bh * S_ + s) * D_ + d] = bv;
          else                 Vt[(bh * D_ + d) * S_ + s] = bv;  // V transposed
        }
    }
  } else {
    float bi[4];
#pragma unroll
    for (int tj = 0; tj < 4; ++tj) bi[tj] = bias[n0 + tj * 16 + nlo];
#pragma unroll
    for (int ti = 0; ti < 2; ++ti)
#pragma unroll
      for (int r = 0; r < 8; ++r) {
        int m = m0 + ti * 16 + r + rofs;
        size_t rowb = (size_t)m * Ndim + n0 + nlo;
#pragma unroll
        for (int tj = 0; tj < 4; ++tj) {
          size_t idx = rowb + tj * 16;
          float v = acc[ti][tj][r] + bi[tj];
          if (MODE == 1) {
            outF[idx] = v + resid[idx];
          } else {
            v = 0.5f * v * (1.0f + erff(v * 0.70710678118654752f));
            outB[idx] = (bf16)v;
          }
        }
      }
  }
}

// ---------------------------------------------------------------------------
// Flash attention: grid (S/64 q-tiles, B*H). 128 threads = 4 waves; each wave
// owns 16 query rows x full D=64. Per 64-key block: S = Q@K^T via 8 WMMAs,
// online softmax (half-wave shfl reductions match the C-layout), P staged in
// LDS (acc-layout -> A-fragment transpose), O += P@V via 8 WMMAs on V^T.
// ---------------------------------------------------------------------------
__global__ void __launch_bounds__(128)
flash_attn_kernel(const bf16* __restrict__ Q, const bf16* __restrict__ K,
                  const bf16* __restrict__ Vt, bf16* __restrict__ O) {
  __shared__ __align__(16) __bf16 lds_p[4][16][72];  // +8 pad: row stride 144B
  int wave = threadIdx.x >> 5;
  int lane = threadIdx.x & 31;
  int qt = blockIdx.x;
  int bh = blockIdx.y;
  const bf16* Qb = Q  + (size_t)bh * S_ * D_;
  const bf16* Kb = K  + (size_t)bh * S_ * D_;
  const bf16* Vb = Vt + (size_t)bh * D_ * S_;
  int q0 = qt * 64 + wave * 16;
  int rofs = (lane >> 4) << 3;
  int nlo  = lane & 15;

  v8f o_acc[4] = {};
  float m_i[8], l_i[8];
#pragma unroll
  for (int r = 0; r < 8; ++r) { m_i[r] = -3.0e38f; l_i[r] = 0.f; }

  for (int kb = 0; kb < S_ / 64; ++kb) {
    int key0 = kb * 64;
    // ---- scores: 16x64 tile of Q @ K^T (contraction over D=64, 2 k-steps)
    v8f s_acc[4] = {};
#pragma unroll
    for (int kk = 0; kk < 64; kk += 32) {
      v16bf aq = load_a_frag(Qb, D_, q0, kk);
#pragma unroll
      for (int j = 0; j < 4; ++j) {
        v16bf bk = load_b_frag(Kb, D_, key0 + j * 16, kk);
        s_acc[j] = wmma_bf16(aq, bk, s_acc[j]);
      }
    }
#pragma unroll
    for (int j = 0; j < 4; ++j)
#pragma unroll
      for (int r = 0; r < 8; ++r) s_acc[j][r] *= 0.125f;  // D^-0.5

    // ---- online softmax: row max over 64 keys (4 accs x 16 lanes/half)
    float mnew[8];
#pragma unroll
    for (int r = 0; r < 8; ++r) {
      float t = fmaxf(fmaxf(s_acc[0][r], s_acc[1][r]),
                      fmaxf(s_acc[2][r], s_acc[3][r]));
      t = fmaxf(t, __shfl_xor(t, 1));
      t = fmaxf(t, __shfl_xor(t, 2));
      t = fmaxf(t, __shfl_xor(t, 4));
      t = fmaxf(t, __shfl_xor(t, 8));
      mnew[r] = fmaxf(m_i[r], t);
    }
    float lsum[8];
#pragma unroll
    for (int r = 0; r < 8; ++r) lsum[r] = 0.f;
#pragma unroll
    for (int j = 0; j < 4; ++j)
#pragma unroll
      for (int r = 0; r < 8; ++r) {
        float p = __expf(s_acc[j][r] - mnew[r]);
        lsum[r] += p;
        lds_p[wave][r + rofs][j * 16 + nlo] = (bf16)p;  // acc-layout store
      }
#pragma unroll
    for (int r = 0; r < 8; ++r) {
      float t = lsum[r];
      t += __shfl_xor(t, 1);
      t += __shfl_xor(t, 2);
      t += __shfl_xor(t, 4);
      t += __shfl_xor(t, 8);
      float corr = __expf(m_i[r] - mnew[r]);
      l_i[r] = l_i[r] * corr + t;
      m_i[r] = mnew[r];
#pragma unroll
      for (int j = 0; j < 4; ++j) o_acc[j][r] *= corr;
    }
    __builtin_amdgcn_wave_barrier();  // LDS in-order within wave; pin schedule

    // ---- O += P @ V  (contraction over 64 keys; V^T gives contiguous B-frags)
#pragma unroll
    for (int kk = 0; kk < 64; kk += 32) {
      const __bf16* pp = &lds_p[wave][nlo][kk + rofs];  // A-frag layout read
      Frag pf;
      pf.q[0] = *(const u32x4*)pp;
      pf.q[1] = *(const u32x4*)(pp + 16);
#pragma unroll
      for (int j = 0; j < 4; ++j) {
        v16bf bv = load_b_frag(Vb, S_, j * 16, key0 + kk);
        o_acc[j] = wmma_bf16(pf.v, bv, o_acc[j]);
      }
    }
  }

  // ---- epilogue: O /= l, write attn_out[B,S,E] bf16 (E col = h*64 + d)
  int b = bh >> 4, h = bh & 15;
#pragma unroll
  for (int j = 0; j < 4; ++j)
#pragma unroll
    for (int r = 0; r < 8; ++r) {
      int srow = q0 + r + rofs;
      int d = j * 16 + nlo;
      float o = o_acc[j][r] / l_i[r];
      O[((size_t)b * S_ + srow) * E_ + h * D_ + d] = (bf16)o;
    }
}

// ---------------------------------------------------------------------------
// Host orchestration
// ---------------------------------------------------------------------------
extern "C" void kernel_launch(void* const* d_in, const int* in_sizes, int n_in,
                              void* d_out, int out_size, void* d_ws, size_t ws_size,
                              hipStream_t stream) {
  (void)in_sizes; (void)n_in; (void)out_size; (void)ws_size;
  const float* x     = (const float*)d_in[0];
  const float* qkv_w = (const float*)d_in[1];
  const float* fc_w  = (const float*)d_in[2];
  const float* fc_b  = (const float*)d_in[3];
  const float* ln1_g = (const float*)d_in[4];
  const float* ln1_b = (const float*)d_in[5];
  const float* ln2_g = (const float*)d_in[6];
  const float* ln2_b = (const float*)d_in[7];
  const float* w1    = (const float*)d_in[8];
  const float* b1    = (const float*)d_in[9];
  const float* w2    = (const float*)d_in[10];
  const float* b2    = (const float*)d_in[11];
  float* out = (float*)d_out;

  char* ws = (char*)d_ws;
  size_t off = 0;
  auto take = [&](size_t bytes) -> void* {
    void* p = ws + off;
    off += (bytes + 255) & ~(size_t)255;
    return p;
  };
  bf16* qkvw_b = (bf16*)take((size_t)3 * E_ * E_ * 2);
  bf16* fcw_b  = (bf16*)take((size_t)E_ * E_ * 2);
  bf16* w1_b   = (bf16*)take((size_t)MLP_ * E_ * 2);
  bf16* w2_b   = (bf16*)take((size_t)E_ * MLP_ * 2);
  bf16* hA     = (bf16*)take((size_t)M_ * E_ * 2);      // ln1(x) bf16
  bf16* Qd     = (bf16*)take((size_t)M_ * E_ * 2);      // [B,H,S,D]
  bf16* Kd     = (bf16*)take((size_t)M_ * E_ * 2);      // [B,H,S,D]
  bf16* Vt     = (bf16*)take((size_t)M_ * E_ * 2);      // [B,H,D,S]
  bf16* attn_o = (bf16*)take((size_t)M_ * E_ * 2);      // [B*S, E]
  float* x1    = (float*)take((size_t)M_ * E_ * 4);     // post-attn residual
  bf16* h2     = (bf16*)take((size_t)M_ * E_ * 2);      // ln2(x1) bf16
  bf16* hmid   = (bf16*)take((size_t)M_ * MLP_ * 2);    // gelu(h2@w1^T+b1)

  // 1) weights -> bf16
  {
    int n;
    n = 3 * E_ * E_;  cvt_bf16_kernel<<<n / 1024, 256, 0, stream>>>(qkv_w, qkvw_b, n);
    n = E_ * E_;      cvt_bf16_kernel<<<n / 1024, 256, 0, stream>>>(fc_w, fcw_b, n);
    n = MLP_ * E_;    cvt_bf16_kernel<<<n / 1024, 256, 0, stream>>>(w1, w1_b, n);
    n = E_ * MLP_;    cvt_bf16_kernel<<<n / 1024, 256, 0, stream>>>(w2, w2_b, n);
  }
  // 2) LN1
  layernorm_kernel<<<M_, 256, 0, stream>>>(x, ln1_g, ln1_b, hA);
  // 3) QKV projection, scatter epilogue (V stored transposed)
  gemm_kernel<0><<<dim3(3 * E_ / 128, M_ / 64), 128, 0, stream>>>(
      hA, qkvw_b, 3 * E_, E_, nullptr, nullptr, nullptr, nullptr, Qd, Kd, Vt);
  // 4) flash attention
  flash_attn_kernel<<<dim3(S_ / 64, B_ * H_), 128, 0, stream>>>(Qd, Kd, Vt, attn_o);
  // 5) output projection + bias + residual
  gemm_kernel<1><<<dim3(E_ / 128, M_ / 64), 128, 0, stream>>>(
      attn_o, fcw_b, E_, E_, fc_b, x, x1, nullptr, nullptr, nullptr, nullptr);
  // 6) LN2
  layernorm_kernel<<<M_, 256, 0, stream>>>(x1, ln2_g, ln2_b, h2);
  // 7) MLP up + GELU
  gemm_kernel<2><<<dim3(MLP_ / 128, M_ / 64), 128, 0, stream>>>(
      h2, w1_b, MLP_, E_, b1, nullptr, nullptr, hmid, nullptr, nullptr, nullptr);
  // 8) MLP down + bias + residual -> d_out
  gemm_kernel<1><<<dim3(E_ / 128, M_ / 64), 128, 0, stream>>>(
      hmid, w2_b, E_, MLP_, b2, x1, out, nullptr, nullptr, nullptr, nullptr);
}
